// FinalModel_39393440039032
// MI455X (gfx1250) — matrix operand order repaired and checked by reading, not compiled, not run on hardware
//
#include <hip/hip_runtime.h>
#include <hip/hip_bf16.h>
#include <stdint.h>

typedef __attribute__((ext_vector_type(16))) _Float16 v16h;
typedef __attribute__((ext_vector_type(8)))  float    v8f;

#define NHID 32
#define SL0  365
#define SL1  366
#define LTOT 732          // 365 + 366 + 1
#define LPAD 736          // padded to multiple of 16 (pad rows of X are zero)
#define NTILE 46          // LPAD / 16
#define NWAVE 8
#define XTS  746          // XT row stride in halves (373 dwords, coprime w/ 64 banks)
#define LN_EPS 1e-5f
#define PI_F 3.14159265358979f

union Frag {
    v16h v;
    uint32_t u[8];
    _Float16 h[16];
};

struct __align__(16) Smem {
    _Float16 X[LPAD * 32];        // xIn (f16), pad rows zero
    _Float16 XT[32 * XTS];        // transposed xIn: XT[c*XTS + t], pads zero
    _Float16 buf[LPAD * 32];      // hidden / a / h1 / h2 / h3 / r (reused in-place)
    // 32x32 weights, stored row-major [out][in] as f16 (B-fragment reads W^T)
    _Float16 w_e0w2[1024], w_e1w2[1024], w_ecw2[1024];
    _Float16 w_q[1024], w_k[1024], w_v[1024], w_o[1024];
    _Float16 w_f1a[1024], w_f1b[1024], w_f2a[1024];
    _Float16 Mt[1024];            // fused attention matrix, [o][c]
    float e0w1[32 * 8];
    float e1w1[32 * 16];
    float ecw1[32 * 32];
    float b_e0b1[32], b_e0b2[32], b_e1b1[32], b_e1b2[32], b_ecb1[32], b_ecb2[32];
    float b_f1a[32], b_f1b[32], b_f2a[32];
    float v_f2w2[32];
    float ln1g[32], ln1b[32], ln2g[32], ln2b[32];
    float pos[LPAD];
    float G[1024];                // Gram = X^T X
    float T[1024];                // G * wk^T
    float S[1024];                // score -> att (softmaxed in place)
    float T1[1024];               // att^T-combined with wo
    float ys[LPAD];
    float red[256];
    float f2b2s;
};

__device__ __forceinline__ v8f wmma16(const Frag& a, const Frag& b, v8f c) {
    return __builtin_amdgcn_wmma_f32_16x16x32_f16(false, a.v, false, b.v,
                                                  (short)0, c, false, false);
}

// A fragment: 16x32 f16, A[m][k] = S[(r0+m)*32 + k]. K pairs are contiguous.
__device__ __forceinline__ void loadA(Frag& f, const _Float16* S, int r0, int lane) {
    int r = r0 + (lane & 15);
    int ko = (lane & 16) ? 8 : 0;
    const _Float16* row = S + r * 32;
#pragma unroll
    for (int j = 0; j < 8; ++j) {
        int k0 = ((j < 4) ? 2 * j : 16 + 2 * (j - 4)) + ko;
        f.u[j] = *(const uint32_t*)(row + k0);
    }
}

// Row-masked A fragment: rows outside [lo,hi) contribute zeros.
// Loads unconditionally (address always in-bounds) and masks the VALUE so the
// compiler emits plain b128 loads + v_cndmask instead of exec-predicated loads.
__device__ __forceinline__ void loadA_masked(Frag& f, const _Float16* S, int r0,
                                             int lane, int lo, int hi) {
    int r = r0 + (lane & 15);
    bool ok = (r >= lo) && (r < hi);
    int ko = (lane & 16) ? 8 : 0;
    const _Float16* row = S + r * 32;
    uint32_t tmp[8];
#pragma unroll
    for (int j = 0; j < 8; ++j) {
        int k0 = ((j < 4) ? 2 * j : 16 + 2 * (j - 4)) + ko;
        tmp[j] = *(const uint32_t*)(row + k0);
    }
#pragma unroll
    for (int j = 0; j < 8; ++j) f.u[j] = ok ? tmp[j] : 0u;
}

// B fragment: 32x16 f16, B[k][n] = W[(o0+n)*32 + k]  (i.e. multiplies by W^T)
__device__ __forceinline__ void loadB(Frag& f, const _Float16* W, int o0, int lane) {
    int n = lane & 15;
    int ko = (lane & 16) ? 8 : 0;
    const _Float16* row = W + (o0 + n) * 32;
#pragma unroll
    for (int j = 0; j < 8; ++j)
        f.u[j] = *(const uint32_t*)(row + ((j < 4) ? 2 * j : 16 + 2 * (j - 4)) + ko);
}

// Gram A fragment from XT: A[m][k] = X[t0+k][h0+m] = XT[(h0+m)*XTS + t0+k]
__device__ __forceinline__ void loadAT(Frag& f, const _Float16* XT, int t0, int h0,
                                       int lane) {
    const _Float16* row = XT + (h0 + (lane & 15)) * XTS + t0;
    int ko = (lane & 16) ? 8 : 0;
#pragma unroll
    for (int j = 0; j < 8; ++j) {
        int k0 = ((j < 4) ? 2 * j : 16 + 2 * (j - 4)) + ko;
        f.u[j] = *(const uint32_t*)(row + k0);
    }
}

// Gram B fragment from XT: B[k][n] = X[t0+k][g0+n] = XT[(g0+n)*XTS + t0+k]
__device__ __forceinline__ void loadBT(Frag& f, const _Float16* XT, int t0, int g0,
                                       int lane) {
    const _Float16* row = XT + (g0 + (lane & 15)) * XTS + t0;
    int ko = (lane & 16) ? 8 : 0;
#pragma unroll
    for (int j = 0; j < 8; ++j)
        f.u[j] = *(const uint32_t*)(row + ((j < 4) ? 2 * j : 16 + 2 * (j - 4)) + ko);
}

// Generic per-tile layer: dst = act(src @ W^T + bias). src may alias dst
// (each wave only touches its own tile's rows; same-wave LDS ops are in-order).
__device__ __forceinline__ void layer_tiles(const _Float16* src, _Float16* dst,
                                            const _Float16* W, const float* bias,
                                            bool relu, int wave, int lane) {
    for (int t = wave; t < NTILE; t += NWAVE) {
        int r0 = t * 16;
        Frag A;  loadA(A, src, r0, lane);
        Frag B0; loadB(B0, W, 0, lane);
        Frag B1; loadB(B1, W, 16, lane);
        v8f acc0 = {}; v8f acc1 = {};
        acc0 = wmma16(A, B0, acc0);
        acc1 = wmma16(A, B1, acc1);
        int c  = lane & 15;
        int rb = r0 + ((lane & 16) ? 8 : 0);
        float b0 = bias ? bias[c] : 0.f;
        float b1 = bias ? bias[16 + c] : 0.f;
#pragma unroll
        for (int v = 0; v < 8; ++v) {
            float x0 = acc0[v] + b0;
            float x1 = acc1[v] + b1;
            if (relu) { x0 = fmaxf(x0, 0.f); x1 = fmaxf(x1, 0.f); }
            dst[(rb + v) * 32 + c]      = (_Float16)x0;
            dst[(rb + v) * 32 + 16 + c] = (_Float16)x1;
        }
    }
}

__device__ __forceinline__ void layernorm_rows(_Float16* a, const _Float16* X,
                                               const float* g, const float* b,
                                               int tid) {
    for (int t = tid; t < LTOT; t += 256) {
        float v[32];
        float m = 0.f;
#pragma unroll
        for (int c = 0; c < 32; ++c) { v[c] = (float)a[t * 32 + c] + (float)X[t * 32 + c]; m += v[c]; }
        m *= (1.f / 32.f);
        float var = 0.f;
#pragma unroll
        for (int c = 0; c < 32; ++c) { float d = v[c] - m; var += d * d; }
        var *= (1.f / 32.f);
        float inv = __frsqrt_rn(var + LN_EPS);
#pragma unroll
        for (int c = 0; c < 32; ++c)
            a[t * 32 + c] = (_Float16)((v[c] - m) * inv * g[c] + b[c]);
    }
}

__device__ __forceinline__ void cvt_copy(_Float16* dst, const float* src, int n, int tid) {
    for (int i = tid; i < n; i += 256) dst[i] = (_Float16)src[i];
}
__device__ __forceinline__ void f32_copy(float* dst, const float* src, int n, int tid) {
    for (int i = tid; i < n; i += 256) dst[i] = src[i];
}

__device__ __forceinline__ float posenc(float p, int c) {
    int i = c >> 1;
    float ang = p * (PI_F / (float)(i + 1));
    return (c & 1) ? __cosf(ang) : __sinf(ang);
}

__global__ void __launch_bounds__(256, 1)
fused_seqmodel(const float* __restrict__ x0, const float* __restrict__ x1,
               const float* __restrict__ pos0, const float* __restrict__ pos1,
               const float* __restrict__ xcT,
               const float* e0w1, const float* e0b1, const float* e0w2, const float* e0b2,
               const float* e1w1, const float* e1b1, const float* e1w2, const float* e1b2,
               const float* ecw1, const float* ecb1, const float* ecw2, const float* ecb2,
               const float* wq, const float* wk, const float* wv, const float* wo,
               const float* ln1g, const float* ln1b, const float* ln2g, const float* ln2b,
               const float* f1w1, const float* f1b1, const float* f1w2, const float* f1b2,
               const float* f2w1, const float* f2b1, const float* f2w2, const float* f2b2,
               float* __restrict__ out) {
    __shared__ Smem sm;
    const int tid  = threadIdx.x;
    const int lane = tid & 31;
    const int wave = tid >> 5;
    const int b    = blockIdx.x;

    // ---- stage weights / params into LDS ----
    cvt_copy(sm.w_e0w2, e0w2, 1024, tid);
    cvt_copy(sm.w_e1w2, e1w2, 1024, tid);
    cvt_copy(sm.w_ecw2, ecw2, 1024, tid);
    cvt_copy(sm.w_q, wq, 1024, tid);
    cvt_copy(sm.w_k, wk, 1024, tid);
    cvt_copy(sm.w_v, wv, 1024, tid);
    cvt_copy(sm.w_o, wo, 1024, tid);
    cvt_copy(sm.w_f1a, f1w1, 1024, tid);
    cvt_copy(sm.w_f1b, f1w2, 1024, tid);
    cvt_copy(sm.w_f2a, f2w1, 1024, tid);
    f32_copy(sm.e0w1, e0w1, 256, tid);
    f32_copy(sm.e1w1, e1w1, 512, tid);
    f32_copy(sm.ecw1, ecw1, 1024, tid);
    f32_copy(sm.b_e0b1, e0b1, 32, tid); f32_copy(sm.b_e0b2, e0b2, 32, tid);
    f32_copy(sm.b_e1b1, e1b1, 32, tid); f32_copy(sm.b_e1b2, e1b2, 32, tid);
    f32_copy(sm.b_ecb1, ecb1, 32, tid); f32_copy(sm.b_ecb2, ecb2, 32, tid);
    f32_copy(sm.b_f1a, f1b1, 32, tid);  f32_copy(sm.b_f1b, f1b2, 32, tid);
    f32_copy(sm.b_f2a, f2b1, 32, tid);  f32_copy(sm.v_f2w2, f2w2, 32, tid);
    f32_copy(sm.ln1g, ln1g, 32, tid);   f32_copy(sm.ln1b, ln1b, 32, tid);
    f32_copy(sm.ln2g, ln2g, 32, tid);   f32_copy(sm.ln2b, ln2b, 32, tid);
    if (tid == 0) sm.f2b2s = f2b2[0];
    for (int t = tid; t < SL0; t += 256) sm.pos[t] = pos0[(size_t)b * SL0 + t];
    for (int t = tid; t < SL1; t += 256) sm.pos[SL0 + t] = pos1[(size_t)b * SL1 + t];
    for (int i = tid; i < 1024; i += 256) sm.G[i] = 0.f;
    __syncthreads();

    // ---- phase 1a: embedding layer-1 (one token per thread), ReLU -> buf ----
    for (int t = tid; t < LTOT; t += 256) {
        if (t < SL0) {
            float xr[8];
            const float* gx = x0 + ((size_t)b * SL0 + t) * 8;
#pragma unroll
            for (int kk = 0; kk < 8; ++kk) xr[kk] = gx[kk];
            for (int j = 0; j < 32; ++j) {
                float acc = sm.b_e0b1[j];
#pragma unroll
                for (int kk = 0; kk < 8; ++kk) acc += sm.e0w1[j * 8 + kk] * xr[kk];
                sm.buf[t * 32 + j] = (_Float16)fmaxf(acc, 0.f);
            }
        } else if (t < SL0 + SL1) {
            float xr[16];
            const float* gx = x1 + ((size_t)b * SL1 + (t - SL0)) * 16;
#pragma unroll
            for (int kk = 0; kk < 16; ++kk) xr[kk] = gx[kk];
            for (int j = 0; j < 32; ++j) {
                float acc = sm.b_e1b1[j];
#pragma unroll
                for (int kk = 0; kk < 16; ++kk) acc += sm.e1w1[j * 16 + kk] * xr[kk];
                sm.buf[t * 32 + j] = (_Float16)fmaxf(acc, 0.f);
            }
        } else {
            float xr[32];
            const float* gx = xcT + (size_t)b * 32;
#pragma unroll
            for (int kk = 0; kk < 32; ++kk) xr[kk] = gx[kk];
            for (int j = 0; j < 32; ++j) {
                float acc = sm.b_ecb1[j];
#pragma unroll
                for (int kk = 0; kk < 32; ++kk) acc += sm.ecw1[j * 32 + kk] * xr[kk];
                sm.buf[t * 32 + j] = (_Float16)fmaxf(acc, 0.f);
            }
        }
    }
    __syncthreads();

    // ---- phase 1b: embedding layer-2 (WMMA) + bias + posenc -> X and XT ----
    for (int t = wave; t < NTILE; t += NWAVE) {
        int r0 = t * 16;
        v8f a0 = {}, a1 = {};
        bool pure0 = (r0 + 16 <= SL0);
        bool pure1 = (r0 >= SL0) && (r0 + 16 <= SL0 + SL1);
        if (pure0 || pure1) {                // fast path: single segment, no mask
            const _Float16* W = pure0 ? sm.w_e0w2 : sm.w_e1w2;
            Frag A;  loadA(A, sm.buf, r0, lane);
            Frag B0; loadB(B0, W, 0, lane);
            Frag B1; loadB(B1, W, 16, lane);
            a0 = wmma16(A, B0, a0); a1 = wmma16(A, B1, a1);
        } else {                             // boundary tiles (22 and 45)
            if (r0 < SL0) {
                Frag A;  loadA_masked(A, sm.buf, r0, lane, 0, SL0);
                Frag B0; loadB(B0, sm.w_e0w2, 0, lane);
                Frag B1; loadB(B1, sm.w_e0w2, 16, lane);
                a0 = wmma16(A, B0, a0); a1 = wmma16(A, B1, a1);
            }
            if (r0 + 16 > SL0 && r0 < SL0 + SL1) {
                Frag A;  loadA_masked(A, sm.buf, r0, lane, SL0, SL0 + SL1);
                Frag B0; loadB(B0, sm.w_e1w2, 0, lane);
                Frag B1; loadB(B1, sm.w_e1w2, 16, lane);
                a0 = wmma16(A, B0, a0); a1 = wmma16(A, B1, a1);
            }
            if (r0 + 16 > SL0 + SL1) {
                Frag A;  loadA_masked(A, sm.buf, r0, lane, SL0 + SL1, LTOT);
                Frag B0; loadB(B0, sm.w_ecw2, 0, lane);
                Frag B1; loadB(B1, sm.w_ecw2, 16, lane);
                a0 = wmma16(A, B0, a0); a1 = wmma16(A, B1, a1);
            }
        }
        int c  = lane & 15;
        int rb = r0 + ((lane & 16) ? 8 : 0);
#pragma unroll
        for (int v = 0; v < 8; ++v) {
            int r = rb + v;
            _Float16 h0, h1;
            if (r < LTOT) {
                const float* b2 = (r < SL0) ? sm.b_e0b2
                                : (r < SL0 + SL1) ? sm.b_e1b2 : sm.b_ecb2;
                float p0 = 0.f, p1 = 0.f;
                if (r < SL0 + SL1) {
                    float p = sm.pos[r];
                    p0 = posenc(p, c);
                    p1 = posenc(p, 16 + c);
                }
                h0 = (_Float16)(a0[v] + b2[c] + p0);
                h1 = (_Float16)(a1[v] + b2[16 + c] + p1);
            } else {                         // zero pad rows (needed for Gram)
                h0 = (_Float16)0.f;
                h1 = (_Float16)0.f;
            }
            sm.X[r * 32 + c]       = h0;
            sm.X[r * 32 + 16 + c]  = h1;
            sm.XT[c * XTS + r]        = h0;   // transposed copy for Gram phase
            sm.XT[(16 + c) * XTS + r] = h1;
        }
    }
    __syncthreads();

    // ---- Gram matrix G = X^T X via WMMA (contract over tokens, reads XT) ----
    {
        v8f g00 = {}, g01 = {}, g10 = {}, g11 = {};
        for (int t = wave; t < NTILE; t += NWAVE) {
            int t0 = t * 16;
            Frag A0, A1, B0, B1;
            loadAT(A0, sm.XT, t0, 0, lane);
            loadAT(A1, sm.XT, t0, 16, lane);
            loadBT(B0, sm.XT, t0, 0, lane);
            loadBT(B1, sm.XT, t0, 16, lane);
            g00 = wmma16(A0, B0, g00);
            g01 = wmma16(A0, B1, g01);
            g10 = wmma16(A1, B0, g10);
            g11 = wmma16(A1, B1, g11);
        }
        int cN = lane & 15;
        int rB = (lane & 16) ? 8 : 0;
#pragma unroll
        for (int v = 0; v < 8; ++v) {
            atomicAdd(&sm.G[(rB + v) * 32 + cN],           g00[v]);
            atomicAdd(&sm.G[(rB + v) * 32 + 16 + cN],      g01[v]);
            atomicAdd(&sm.G[(16 + rB + v) * 32 + cN],      g10[v]);
            atomicAdd(&sm.G[(16 + rB + v) * 32 + 16 + cN], g11[v]);
        }
    }
    __syncthreads();

    // ---- score = wq * G * wk^T / sqrt(L); softmax; M = wv^T att^T wo^T ----
    for (int i = tid; i < 1024; i += 256) {        // T[a][g] = sum_q G[a][q] * wk[g][q]
        int a_ = i >> 5, g_ = i & 31;
        float s = 0.f;
        for (int q = 0; q < 32; ++q) s += sm.G[a_ * 32 + q] * (float)sm.w_k[g_ * 32 + q];
        sm.T[i] = s;
    }
    __syncthreads();
    {
        const float sscale = 0.036961105f;         // 1/sqrt(732)
        for (int i = tid; i < 1024; i += 256) {    // S[h][g] = sum_a wq[h][a] * T[a][g]
            int h_ = i >> 5, g_ = i & 31;
            float s = 0.f;
            for (int q = 0; q < 32; ++q) s += (float)sm.w_q[h_ * 32 + q] * sm.T[q * 32 + g_];
            sm.S[i] = s * sscale;
        }
    }
    __syncthreads();
    if (tid < 32) {                                // softmax over g, in place
        float mx = -3.4e38f;
        for (int g = 0; g < 32; ++g) mx = fmaxf(mx, sm.S[tid * 32 + g]);
        float sum = 0.f;
        for (int g = 0; g < 32; ++g) {
            float e = __expf(sm.S[tid * 32 + g] - mx);
            sm.S[tid * 32 + g] = e;
            sum += e;
        }
        float inv = 1.f / sum;
        for (int g = 0; g < 32; ++g) sm.S[tid * 32 + g] *= inv;
    }
    __syncthreads();
    for (int i = tid; i < 1024; i += 256) {        // T1[g][o] = sum_h att[h][g] * wo[o][h]
        int g_ = i >> 5, o_ = i & 31;
        float s = 0.f;
        for (int h = 0; h < 32; ++h) s += sm.S[h * 32 + g_] * (float)sm.w_o[o_ * 32 + h];
        sm.T1[g_ * 32 + o_] = s;
    }
    __syncthreads();
    for (int i = tid; i < 1024; i += 256) {        // Mt[o][c] = sum_g wv[g][c] * T1[g][o]
        int o_ = i >> 5, c_ = i & 31;
        float s = 0.f;
        for (int g = 0; g < 32; ++g) s += (float)sm.w_v[g * 32 + c_] * sm.T1[g * 32 + o_];
        sm.Mt[o_ * 32 + c_] = (_Float16)s;
    }
    __syncthreads();

    // ---- a = X @ M  (attention output + wo, fused) ----
    layer_tiles(sm.X, sm.buf, sm.Mt, nullptr, false, wave, lane);
    __syncthreads();
    // ---- h1 = LN(a + X) ----
    layernorm_rows(sm.buf, sm.X, sm.ln1g, sm.ln1b, tid);
    __syncthreads();
    // ---- ffn1: t = ReLU(h1 W1^T + b); h2 = t W2^T + b ----
    layer_tiles(sm.buf, sm.buf, sm.w_f1a, sm.b_f1a, true, wave, lane);
    __syncthreads();
    layer_tiles(sm.buf, sm.buf, sm.w_f1b, sm.b_f1b, false, wave, lane);
    __syncthreads();
    // ---- h3 = LN(h2 + X) ----
    layernorm_rows(sm.buf, sm.X, sm.ln2g, sm.ln2b, tid);
    __syncthreads();
    // ---- ffn2 layer-1: r = ReLU(h3 W^T + b) ----
    layer_tiles(sm.buf, sm.buf, sm.w_f2a, sm.b_f2a, true, wave, lane);
    __syncthreads();
    // ---- head: y[t] = r[t] . f2w2 + f2b2 ----
    for (int t = tid; t < LTOT; t += 256) {
        float s = sm.f2b2s;
#pragma unroll
        for (int c = 0; c < 32; ++c) s += (float)sm.buf[t * 32 + c] * sm.v_f2w2[c];
        sm.ys[t] = s;
    }
    __syncthreads();
    // ---- segment-mean pooling + block reduce ----
    float part = 0.f;
    for (int t = tid; t < LTOT; t += 256) {
        float w = (t < SL0) ? (1.f / 365.f) : (t < SL0 + SL1) ? (1.f / 366.f) : 1.f;
        part += sm.ys[t] * w;
    }
    sm.red[tid] = part;
    for (int s = 128; s > 0; s >>= 1) {
        __syncthreads();
        if (tid < s) sm.red[tid] += sm.red[tid + s];
    }
    if (tid == 0) out[b] = sm.red[0];
}

extern "C" void kernel_launch(void* const* d_in, const int* in_sizes, int n_in,
                              void* d_out, int out_size, void* d_ws, size_t ws_size,
                              hipStream_t stream) {
    (void)n_in; (void)d_ws; (void)ws_size; (void)out_size;
    const float* x0   = (const float*)d_in[0];
    const float* x1   = (const float*)d_in[1];
    const float* pos0 = (const float*)d_in[2];
    const float* pos1 = (const float*)d_in[3];
    const float* xcT  = (const float*)d_in[4];
    const float* e0w1 = (const float*)d_in[5];
    const float* e0b1 = (const float*)d_in[6];
    const float* e0w2 = (const float*)d_in[7];
    const float* e0b2 = (const float*)d_in[8];
    const float* e1w1 = (const float*)d_in[9];
    const float* e1b1 = (const float*)d_in[10];
    const float* e1w2 = (const float*)d_in[11];
    const float* e1b2 = (const float*)d_in[12];
    const float* ecw1 = (const float*)d_in[13];
    const float* ecb1 = (const float*)d_in[14];
    const float* ecw2 = (const float*)d_in[15];
    const float* ecb2 = (const float*)d_in[16];
    const float* wq   = (const float*)d_in[17];
    const float* wk   = (const float*)d_in[18];
    const float* wv   = (const float*)d_in[19];
    const float* wo   = (const float*)d_in[20];
    const float* ln1g = (const float*)d_in[21];
    const float* ln1b = (const float*)d_in[22];
    const float* ln2g = (const float*)d_in[23];
    const float* ln2b = (const float*)d_in[24];
    const float* f1w1 = (const float*)d_in[25];
    const float* f1b1 = (const float*)d_in[26];
    const float* f1w2 = (const float*)d_in[27];
    const float* f1b2 = (const float*)d_in[28];
    const float* f2w1 = (const float*)d_in[29];
    const float* f2b1 = (const float*)d_in[30];
    const float* f2w2 = (const float*)d_in[31];
    const float* f2b2 = (const float*)d_in[32];
    float* out = (float*)d_out;

    const int B = in_sizes[0] / (SL0 * 8);   // batch from x0 element count

    hipLaunchKernelGGL(fused_seqmodel, dim3(B), dim3(256), 0, stream,
                       x0, x1, pos0, pos1, xcT,
                       e0w1, e0b1, e0w2, e0b2, e1w1, e1b1, e1w2, e1b2,
                       ecw1, ecb1, ecw2, ecb2, wq, wk, wv, wo,
                       ln1g, ln1b, ln2g, ln2b,
                       f1w1, f1b1, f1w2, f1b2, f2w1, f2b1, f2w2, f2b2,
                       out);
}
